// MultiheadMaskedAttention_11338713662119
// MI455X (gfx1250) — matrix operand order repaired and checked by reading, not compiled, not run on hardware
//
#include <hip/hip_runtime.h>

// ---------------------------------------------------------------------------
// MI455X (gfx1250) multi-head causal attention, bf16 WMMA + TDM pipeline.
// ---------------------------------------------------------------------------

typedef __bf16 bf16_t;
typedef __attribute__((ext_vector_type(16))) __bf16 v16bf;
typedef __attribute__((ext_vector_type(8)))  __bf16 v8bf;
typedef __attribute__((ext_vector_type(8)))  float  v8f;

typedef unsigned int u32x4 __attribute__((ext_vector_type(4)));
typedef int          i32x8 __attribute__((ext_vector_type(8)));
typedef int          i32x4 __attribute__((ext_vector_type(4)));

#define HIDDEN 1024
#define HEADS  16
#define HEADD  64
#define SEQ    2048
#define BATCH  4
#define ROWS   (BATCH * SEQ)
#define NEG_INF (-1000000.0f)

static __device__ __forceinline__ v8f wmma_bf16(v16bf a, v16bf b, v8f c) {
  // D = A(16x32 bf16) * B(32x16 bf16) + C(16x16 f32)
  return __builtin_amdgcn_wmma_f32_16x16x32_bf16(
      /*neg_a=*/false, a, /*neg_b=*/false, b,
      /*c_mod=*/(short)0, c, /*reuse_a=*/false, /*reuse_b=*/false);
}

// A-matrix 16x32 bf16 fragment. Lane (li = lane&15) owns row li.
// elems 0..7  -> K = 8*hh + e ; elems 8..15 -> K = 16 + 8*hh + (e-8)
static __device__ __forceinline__ v16bf load_a_frag(const bf16_t* row, int hh) {
  union { v16bf v; v8bf h[2]; } u;
  u.h[0] = *(const v8bf*)(row + 8 * hh);
  u.h[1] = *(const v8bf*)(row + 16 + 8 * hh);
  return u.v;
}

// B-matrix 32x16 bf16 fragment. Lane (li) owns column li.
// elems 0..7  -> K = 8*hh + e ; elems 8..15 -> K = 16 + 8*hh + (e-8)
static __device__ __forceinline__ v16bf load_b_frag(const bf16_t* col, int hh) {
  union { v16bf v; v8bf h[2]; } u;
  u.h[0] = *(const v8bf*)(col + 8 * hh);
  u.h[1] = *(const v8bf*)(col + 16 + 8 * hh);
  return u.v;
}

// ---------------------------------------------------------------------------
// Tensor Data Mover: async 2D tile load Global -> LDS (bf16 elements).
// D# built per CDNA5 ISA 8.3/8.4: group0 = {flags, lds_addr, global_addr,
// type=2}; group1 = {data_size=2B, tensor dims == tile dims, tile dims,
// dim0 stride}; groups 2/3 zero (2D tensor). Tracked by TENSORcnt.
// This toolchain's builtin takes 6 args (extra i32x8 group before cpol).
// ---------------------------------------------------------------------------
static __device__ __forceinline__ void tdm_load_2d(const void* lds_dst,
                                                   const void* gsrc,
                                                   unsigned tile0,   // contiguous dim (elems)
                                                   unsigned tile1,   // rows
                                                   unsigned long long stride0) { // elems
  const unsigned long long ga  = (unsigned long long)gsrc;
  const unsigned lds_addr = (unsigned)(unsigned long long)lds_dst; // low 32 = LDS offset

  u32x4 g0;
  g0[0] = 1u;                                               // count=1, user D#
  g0[1] = lds_addr;                                         // LDS byte address
  g0[2] = (unsigned)(ga & 0xffffffffu);                     // global_addr[31:0]
  g0[3] = (unsigned)((ga >> 32) & 0x01ffffffu) | 0x80000000u; // ga[56:32] | type=2

  i32x8 g1;
  g1[0] = (int)(1u << 16);                                  // data_size=1 (2 bytes)
  g1[1] = (int)((tile0 & 0xffffu) << 16);                   // tensor_dim0[15:0]
  g1[2] = (int)((tile0 >> 16) | ((tile1 & 0xffffu) << 16)); // td0 hi | tensor_dim1 lo
  g1[3] = (int)((tile1 >> 16) | (tile0 << 16));             // td1 hi | tile_dim0
  g1[4] = (int)(tile1 & 0xffffu);                           // tile_dim1 (tile_dim2=0)
  g1[5] = (int)(stride0 & 0xffffffffull);                   // tensor_dim0_stride lo
  g1[6] = (int)((stride0 >> 32) & 0xffffull);               // stride0 hi | stride1 lo=0
  g1[7] = 0;

  const i32x4 z4 = {0, 0, 0, 0};
  const i32x8 z8 = {0, 0, 0, 0, 0, 0, 0, 0};
  __builtin_amdgcn_tensor_load_to_lds(g0, g1, z4, z4, z8, 0);
}

// ---------------------------------------------------------------------------
// fp32 -> bf16 conversion (grid-stride)
// ---------------------------------------------------------------------------
__global__ void cvt_f32_bf16(const float* __restrict__ in, bf16_t* __restrict__ out, int n) {
  for (int i = blockIdx.x * blockDim.x + threadIdx.x; i < n; i += gridDim.x * blockDim.x)
    out[i] = (bf16_t)in[i];
}

// ---------------------------------------------------------------------------
// QKV projection: qkv = x16 @ Wqkv16 + b_qkv, scatter to Q / K / V^T (bf16).
// Block = 128 threads (4 waves) -> 64(M) x 64(N) tile. K stepped by 32.
// (B-tile needs a transpose, which TDM cannot do -> manual LDS staging.)
// ---------------------------------------------------------------------------
__global__ __launch_bounds__(128) void qkv_gemm(const bf16_t* __restrict__ A,
                                                const bf16_t* __restrict__ B,
                                                const float*  __restrict__ bias,
                                                bf16_t* __restrict__ Qo,
                                                bf16_t* __restrict__ Ko,
                                                bf16_t* __restrict__ Vt) {
  __shared__ __align__(32) bf16_t Bt[64][40];  // B tile transposed: Bt[n][k]
  const int n0 = blockIdx.x * 64;
  const int m0 = blockIdx.y * 64;
  const int tid  = threadIdx.x;
  const int wave = tid >> 5;
  const int lane = tid & 31;
  const int li = lane & 15, hh = lane >> 4;
  const int ldb = 3 * HIDDEN;

  const v8f vzero = {0.f, 0.f, 0.f, 0.f, 0.f, 0.f, 0.f, 0.f};
  v8f acc[4] = {vzero, vzero, vzero, vzero};

  const size_t arow = (size_t)(m0 + 16 * wave + li) * HIDDEN;

  for (int k0 = 0; k0 < HIDDEN; k0 += 32) {
    for (int idx = tid; idx < 64 * 32; idx += 128) {
      const int kk = idx & 31, nn = idx >> 5;
      Bt[nn][kk] = B[(size_t)(k0 + kk) * ldb + n0 + nn];
    }
    __syncthreads();
    if (k0 + 32 < HIDDEN) __builtin_prefetch(A + arow + k0 + 32, 0, 3);
    const v16bf af = load_a_frag(A + arow + k0, hh);
#pragma unroll
    for (int t = 0; t < 4; ++t) {
      const v16bf bfrag = load_b_frag(&Bt[16 * t + li][0], hh);
      acc[t] = wmma_bf16(af, bfrag, acc[t]);
    }
    __syncthreads();
  }

#pragma unroll
  for (int t = 0; t < 4; ++t) {
    const int c = n0 + 16 * t + li;
    const float bv = bias[c];
#pragma unroll
    for (int r = 0; r < 8; ++r) {
      const int m = m0 + 16 * wave + r + 8 * hh;
      const bf16_t o = (bf16_t)(acc[t][r] + bv);
      const int bb = m >> 11;           // m / SEQ
      const int ss = m & (SEQ - 1);     // m % SEQ
      if (c < HIDDEN) {
        Qo[(size_t)m * HIDDEN + c] = o;                       // (B,S,h,d)
      } else if (c < 2 * HIDDEN) {
        Ko[(size_t)m * HIDDEN + (c - HIDDEN)] = o;            // (B,S,h,d)
      } else {
        const int cc = c - 2 * HIDDEN;
        const int h = cc >> 6, d = cc & 63;
        Vt[(((size_t)(bb * HEADS + h)) * HEADD + d) * SEQ + ss] = o;  // (B,h,d,S)
      }
    }
  }
}

// ---------------------------------------------------------------------------
// Flash-style causal attention. Block = 128 threads (4 waves) = 64 queries of
// one (b,h); wave w owns queries [qc*64+16w, +16). K/V tiles for each 32-key
// block are staged ONCE per block into LDS by the Tensor Data Mover (issued by
// wave 0, TENSORcnt-tracked), shared by all 4 waves. Online softmax in f32
// registers; P transposed C->A layout through per-wave LDS.
// ---------------------------------------------------------------------------
__global__ __launch_bounds__(128) void attn(const bf16_t* __restrict__ Q,
                                            const bf16_t* __restrict__ K,
                                            const bf16_t* __restrict__ Vt,
                                            bf16_t* __restrict__ O) {
  __shared__ __align__(32) bf16_t Kt[32][64];      // [key][d]  (TDM-loaded)
  __shared__ __align__(32) bf16_t Vs[64][32];      // [d][key]  (TDM-loaded)
  __shared__ __align__(32) bf16_t Pt[4][16][32];   // per-wave P staging

  const int qc = blockIdx.x & (SEQ / 64 - 1);   // 32 query chunks
  const int bh = blockIdx.x >> 5;
  const int h  = bh & (HEADS - 1);
  const int b  = bh / HEADS;
  const int wave = threadIdx.x >> 5;
  const int lane = threadIdx.x & 31;
  const int li = lane & 15, hh = lane >> 4;
  const int qbase = qc * 64 + 16 * wave;
  const float scale = 0.125f;  // 1/sqrt(64)

  // Q fragments for this wave's 16 queries (head dim 64 = 2 x K32)
  const bf16_t* qrow = Q + ((size_t)(b * SEQ + qbase + li) * HEADS + h) * HEADD;
  const v16bf qa0 = load_a_frag(qrow, hh);
  const v16bf qa1 = load_a_frag(qrow + 32, hh);

  const v8f vzero = {0.f, 0.f, 0.f, 0.f, 0.f, 0.f, 0.f, 0.f};
  v8f o[4] = {vzero, vzero, vzero, vzero};
  float m_run[8], l_run[8];
#pragma unroll
  for (int r = 0; r < 8; ++r) { m_run[r] = -3.0e38f; l_run[r] = 0.f; }

  const int qmax = qbase + 15;          // per-wave causal bound
  const int blockQmax = qc * 64 + 63;   // block-wide key range
  const bf16_t* khead = K  + ((size_t)(b * SEQ) * HEADS + h) * HEADD;
  const bf16_t* vhead = Vt + ((size_t)(b * HEADS + h)) * HEADD * SEQ;

  for (int kb = 0; kb <= blockQmax; kb += 32) {
    // ---- wave 0: async TDM stage of K/V tiles for this key block ----------
    if (threadIdx.x < 32) {
      tdm_load_2d(&Kt[0][0], khead + (size_t)kb * HIDDEN, /*d*/ 64, /*keys*/ 32,
                  /*stride*/ (unsigned long long)HEADS * HEADD);
      tdm_load_2d(&Vs[0][0], vhead + kb, /*keys*/ 32, /*d*/ 64,
                  /*stride*/ (unsigned long long)SEQ);
      __builtin_amdgcn_s_wait_tensorcnt(0);
    }
    __syncthreads();

    if (kb <= qmax) {  // uniform per wave: EXEC all-ones inside
      // ---- scores: two 16x16 tiles over this 32-key block -----------------
      v8f s[2];
#pragma unroll
      for (int t = 0; t < 2; ++t) {
        const bf16_t* kp = &Kt[16 * t + li][0];
        const v16bf kb0 = load_b_frag(kp, hh);        // d = 0..31
        const v16bf kb1 = load_b_frag(kp + 32, hh);   // d = 32..63
        v8f z = vzero;
        z = wmma_bf16(qa0, kb0, z);
        z = wmma_bf16(qa1, kb1, z);
        s[t] = z;
      }

      // ---- mask + scale, per-tile row max ---------------------------------
      float mt[8], p0[8], p1[8];
#pragma unroll
      for (int r = 0; r < 8; ++r) {
        const int q = qbase + r + 8 * hh;
        float v0 = s[0][r] * scale; if (kb + li > q)       v0 = NEG_INF;
        float v1 = s[1][r] * scale; if (kb + 16 + li > q)  v1 = NEG_INF;
        s[0][r] = v0; s[1][r] = v1;
        mt[r] = fmaxf(v0, v1);
      }
      for (int off = 1; off < 16; off <<= 1)
#pragma unroll
        for (int r = 0; r < 8; ++r)
          mt[r] = fmaxf(mt[r], __shfl_xor(mt[r], off, 32));

      // ---- online softmax update ------------------------------------------
      float rs[8];
#pragma unroll
      for (int r = 0; r < 8; ++r) {
        const float mn = fmaxf(m_run[r], mt[r]);
        const float alpha = __expf(m_run[r] - mn);
        m_run[r] = mn;
        const float e0 = __expf(s[0][r] - mn);
        const float e1 = __expf(s[1][r] - mn);
        p0[r] = e0; p1[r] = e1;
        rs[r] = e0 + e1;
        l_run[r] = l_run[r] * alpha;
#pragma unroll
        for (int nt = 0; nt < 4; ++nt) o[nt][r] = o[nt][r] * alpha;
      }
      for (int off = 1; off < 16; off <<= 1)
#pragma unroll
        for (int r = 0; r < 8; ++r)
          rs[r] += __shfl_xor(rs[r], off, 32);
#pragma unroll
      for (int r = 0; r < 8; ++r) l_run[r] += rs[r];

      // ---- transpose P (C layout) -> A fragment via per-wave LDS ----------
#pragma unroll
      for (int r = 0; r < 8; ++r) {
        Pt[wave][r + 8 * hh][li]      = (bf16_t)p0[r];
        Pt[wave][r + 8 * hh][16 + li] = (bf16_t)p1[r];
      }
      asm volatile("s_wait_dscnt 0" ::: "memory");  // same-wave LDS RAW
      const v16bf pa = load_a_frag(&Pt[wave][li][0], hh);

      // ---- O += P @ V : 4 N-tiles over head dim ---------------------------
#pragma unroll
      for (int nt = 0; nt < 4; ++nt) {
        const v16bf vf = load_b_frag(&Vs[16 * nt + li][0], hh);
        o[nt] = wmma_bf16(pa, vf, o[nt]);
      }
    }
    __syncthreads();  // protect Kt/Vs before next TDM stage
  }

  // ---- normalize and store bf16 attention output (B,S,h,d) ----------------
#pragma unroll
  for (int nt = 0; nt < 4; ++nt) {
    const int d = 16 * nt + li;
#pragma unroll
    for (int r = 0; r < 8; ++r) {
      const int q = qbase + r + 8 * hh;
      O[((size_t)(b * SEQ + q) * HEADS + h) * HEADD + d] = (bf16_t)(o[nt][r] / l_run[r]);
    }
  }
}

// ---------------------------------------------------------------------------
// Output projection: out = attn16 @ Wo16 + b_o  (fp32 out)
// ---------------------------------------------------------------------------
__global__ __launch_bounds__(128) void out_gemm(const bf16_t* __restrict__ A,
                                                const bf16_t* __restrict__ B,
                                                const float*  __restrict__ bias,
                                                float* __restrict__ C) {
  __shared__ __align__(32) bf16_t Bt[64][40];
  const int n0 = blockIdx.x * 64;
  const int m0 = blockIdx.y * 64;
  const int tid  = threadIdx.x;
  const int wave = tid >> 5;
  const int lane = tid & 31;
  const int li = lane & 15, hh = lane >> 4;

  const v8f vzero = {0.f, 0.f, 0.f, 0.f, 0.f, 0.f, 0.f, 0.f};
  v8f acc[4] = {vzero, vzero, vzero, vzero};
  const size_t arow = (size_t)(m0 + 16 * wave + li) * HIDDEN;

  for (int k0 = 0; k0 < HIDDEN; k0 += 32) {
    for (int idx = tid; idx < 64 * 32; idx += 128) {
      const int kk = idx & 31, nn = idx >> 5;
      Bt[nn][kk] = B[(size_t)(k0 + kk) * HIDDEN + n0 + nn];
    }
    __syncthreads();
    if (k0 + 32 < HIDDEN) __builtin_prefetch(A + arow + k0 + 32, 0, 3);
    const v16bf af = load_a_frag(A + arow + k0, hh);
#pragma unroll
    for (int t = 0; t < 4; ++t) {
      const v16bf bfrag = load_b_frag(&Bt[16 * t + li][0], hh);
      acc[t] = wmma_bf16(af, bfrag, acc[t]);
    }
    __syncthreads();
  }

#pragma unroll
  for (int t = 0; t < 4; ++t) {
    const int c = n0 + 16 * t + li;
    const float bv = bias[c];
#pragma unroll
    for (int r = 0; r < 8; ++r) {
      const int m = m0 + 16 * wave + r + 8 * hh;
      C[(size_t)m * HIDDEN + c] = acc[t][r] + bv;
    }
  }
}

// ---------------------------------------------------------------------------
// Host launch
// ---------------------------------------------------------------------------
extern "C" void kernel_launch(void* const* d_in, const int* in_sizes, int n_in,
                              void* d_out, int out_size, void* d_ws, size_t ws_size,
                              hipStream_t stream) {
  const float* x     = (const float*)d_in[0];   // (4,2048,1024)
  const float* W_qkv = (const float*)d_in[1];   // (1024,3072)
  const float* b_qkv = (const float*)d_in[2];   // (3072,)
  const float* W_o   = (const float*)d_in[3];   // (1024,1024)
  const float* b_o   = (const float*)d_in[4];   // (1024,)
  float* out = (float*)d_out;

  // Workspace layout (bf16 elements), ~88 MB total.
  bf16_t* x16    = (bf16_t*)d_ws;
  bf16_t* Wqkv16 = x16    + (size_t)ROWS * HIDDEN;
  bf16_t* Wo16   = Wqkv16 + (size_t)HIDDEN * 3 * HIDDEN;
  bf16_t* Q16    = Wo16   + (size_t)HIDDEN * HIDDEN;
  bf16_t* K16    = Q16    + (size_t)ROWS * HIDDEN;
  bf16_t* Vt16   = K16    + (size_t)ROWS * HIDDEN;
  bf16_t* AO16   = Vt16   + (size_t)ROWS * HIDDEN;

  cvt_f32_bf16<<<512, 256, 0, stream>>>(x,     x16,    ROWS * HIDDEN);
  cvt_f32_bf16<<<512, 256, 0, stream>>>(W_qkv, Wqkv16, HIDDEN * 3 * HIDDEN);
  cvt_f32_bf16<<<512, 256, 0, stream>>>(W_o,   Wo16,   HIDDEN * HIDDEN);

  qkv_gemm<<<dim3(3 * HIDDEN / 64, ROWS / 64), 128, 0, stream>>>(
      x16, Wqkv16, b_qkv, Q16, K16, Vt16);

  attn<<<BATCH * HEADS * (SEQ / 64), 128, 0, stream>>>(Q16, K16, Vt16, AO16);

  out_gemm<<<dim3(HIDDEN / 64, ROWS / 64), 128, 0, stream>>>(AO16, Wo16, b_o, out);
}
